// HK_Pro_Model_74105365725695
// MI455X (gfx1250) — compile-verified, matrix operand order checked by reading, not run on hardware
//
#include <hip/hip_runtime.h>
#include <hip/hip_bf16.h>

// ---------------------------------------------------------------------------
// Problem constants (fixed by the reference)
// ---------------------------------------------------------------------------
static constexpr int Bb      = 8;
static constexpr int Nn      = 5000;
static constexpr int Tt      = 24;
static constexpr int Ff      = 10;
static constexpr int Hh      = 128;
static constexpr int HEADS   = 4;
static constexpr int HORIZON = 6;
static constexpr int Ee      = 80000;
static constexpr int ROWS    = Bb * Nn;            // 40000
static constexpr int G3H     = 3 * Hh;             // 384
static constexpr int EDGE_TOTAL = Bb * Ee + ROWS;  // 680000
static constexpr float NEG_SLOPE = 0.2f;

typedef __attribute__((ext_vector_type(16))) _Float16 v16h;
typedef __attribute__((ext_vector_type(8)))  float    v8f;

// ---------------------------------------------------------------------------
// Small helpers
// ---------------------------------------------------------------------------
static __device__ inline v8f v8f_zero() {
  v8f z;
#pragma unroll
  for (int i = 0; i < 8; ++i) z[i] = 0.f;
  return z;
}

static __device__ inline v8f wmma_f16(v16h a, v16h b, v8f c) {
  // D = A(16x32 f16) * B(32x16 f16) + C(16x16 f32)
  return __builtin_amdgcn_wmma_f32_16x16x32_f16(
      /*neg_a=*/false, a, /*neg_b=*/false, b,
      /*c_mod=*/(short)0, c, /*reuse_a=*/false, /*reuse_b=*/false);
}

static __device__ inline unsigned pack2h(float a, float b) {
  union { _Float16 h[2]; unsigned u; } r;
  r.h[0] = (_Float16)a; r.h[1] = (_Float16)b;
  return r.u;
}

// Order-preserving float->uint key for atomicMax-based segment max
static __device__ inline unsigned fkey(float f) {
  union { float f; unsigned u; } c; c.f = f;
  return (c.u & 0x80000000u) ? ~c.u : (c.u | 0x80000000u);
}
static __device__ inline float funkey(unsigned u) {
  union { float f; unsigned u; } c;
  c.u = (u & 0x80000000u) ? (u & 0x7fffffffu) : ~u;
  return c.f;
}

// A-fragment (16x32 f16) from an LDS buffer laid out [16 rows][128 cols f16].
// ISA layout: lane<16 -> M=lane, dword d holds K pair at
//   k = kt*32 + (d>>2)*16 + half*8 + (d&3)*2   (dword index = k/2)
static __device__ inline v16h load_afrag(const _Float16* h16, int kt, int lane) {
  const int hv = lane >> 4;        // half
  const int m  = lane & 15;
  const unsigned* row = (const unsigned*)(h16 + (size_t)m * Hh);
  const int base = kt * 16 + hv * 4;
  union { v16h h; unsigned u[8]; } r;
#pragma unroll
  for (int d = 0; d < 8; ++d)
    r.u[d] = row[base + (d >> 2) * 8 + (d & 3)];
  return r.h;
}

// B-fragment: pre-packed in global memory, [frag][lane][8 dwords]
static __device__ inline v16h load_bfrag(const unsigned* __restrict__ packed,
                                         int frag, int lane) {
  const uint4* p = (const uint4*)(packed + ((size_t)frag * 32 + lane) * 8);
  union { v16h h; uint4 q[2]; } r;
  r.q[0] = p[0];
  r.q[1] = p[1];
  return r.h;
}

// Branchless C-scatter: gi region for r/z tiles (nt<16), ghn region for
// n-gate tiles. ghn is pre-zeroed each step so "+=" is correct for both.
static __device__ inline void scatter_c(float* __restrict__ gi,
                                        float* __restrict__ ghn,
                                        int nt, int hv, int lrow, v8f c) {
  const int n = nt * 16 + lrow;
#pragma unroll
  for (int vr = 0; vr < 8; ++vr) {
    int m = vr + hv * 8;
    float* dst = (nt < 16) ? (gi + m * G3H + n) : (ghn + m * Hh + (n - 2 * Hh));
    *dst += c[vr];
  }
}

// ---------------------------------------------------------------------------
// Weight packing: W is row-major [Ncols][K] (we compute x @ W.T), build
// B-fragments with lane = N column, dwords = K pairs (mirror of A layout).
// ---------------------------------------------------------------------------
__global__ __launch_bounds__(256)
void pack_b_kernel(const float* __restrict__ W, unsigned* __restrict__ out,
                   int NT, int KT, int K) {
  int g = blockIdx.x * 256 + threadIdx.x;
  int total = NT * KT * 32;
  if (g >= total) return;
  int lane = g & 31;
  int frag = g >> 5;
  int kt = frag % KT;
  int nt = frag / KT;
  int hv = lane >> 4;
  int n  = nt * 16 + (lane & 15);
  const float* wrow = W + (size_t)n * K;
  unsigned* dst = out + (size_t)frag * 256 + lane * 8;
#pragma unroll
  for (int d = 0; d < 8; ++d) {
    int k = kt * 32 + ((d >> 2) << 4) + hv * 8 + ((d & 3) << 1);
    dst[d] = pack2h(wrow[k], wrow[k + 1]);
  }
}

// ---------------------------------------------------------------------------
// GRU layer 0: x(40000,24,10) -> h1(40000,24,128) f16.
// Block = 256 thr (8 waves) owns 16 rows, loops t internally.
// Recurrent GEMM per step via WMMA; input proj (F=10) via VALU.
// ---------------------------------------------------------------------------
__global__ __launch_bounds__(256)
void gru_layer0_kernel(const float* __restrict__ x,
                       const float* __restrict__ w_ih0,
                       const float* __restrict__ b_ih0,
                       const float* __restrict__ b_hh0,
                       const unsigned* __restrict__ whh_packed,
                       _Float16* __restrict__ h1_out) {
  __shared__ float lds_wih[G3H * Ff];               // 15360 B
  __shared__ float lds_gi[16 * G3H];                // 24576 B
  __shared__ float lds_ghn[16 * Hh];                // 8192 B
  __shared__ float lds_h[16 * Hh];                  // 8192 B
  __shared__ __align__(16) _Float16 lds_h16[16 * Hh]; // 4096 B
  __shared__ float lds_x[16 * Ff];                  // 640 B
  __shared__ float lds_bias[G3H];                   // fused b_ih + b_hh (r,z)
  __shared__ float lds_bhn[Hh];                     // b_hh n-gate part

  const int tid = threadIdx.x;
  const int r0  = blockIdx.x * 16;
  const int wave = tid >> 5, lane = tid & 31, hv = lane >> 4, lrow = lane & 15;

  for (int i = tid; i < G3H * Ff; i += 256) lds_wih[i] = w_ih0[i];
  for (int i = tid; i < 16 * Hh; i += 256) {
    lds_h[i] = 0.f;
    lds_h16[i] = (_Float16)0.f;
  }
  for (int n = tid; n < G3H; n += 256)
    lds_bias[n] = b_ih0[n] + (n < 2 * Hh ? b_hh0[n] : 0.f);
  for (int j = tid; j < Hh; j += 256) lds_bhn[j] = b_hh0[2 * Hh + j];

  for (int t = 0; t < Tt; ++t) {
    __syncthreads();
    // stage x_t
    for (int i = tid; i < 16 * Ff; i += 256) {
      int m = i / Ff, f = i - m * Ff;
      lds_x[i] = x[((size_t)(r0 + m) * Tt + t) * Ff + f];
    }
    // pre-zero n-gate accumulator (enables branchless WMMA scatter)
    for (int i = tid; i < 16 * Hh; i += 256) lds_ghn[i] = 0.f;
    __syncthreads();
    // gi = x_t @ w_ih0.T + fused bias
    for (int p = tid; p < 16 * G3H; p += 256) {
      int m = p / G3H, n = p - m * G3H;
      float acc = lds_bias[n];
#pragma unroll
      for (int f = 0; f < Ff; ++f) acc += lds_x[m * Ff + f] * lds_wih[n * Ff + f];
      lds_gi[p] = acc;
    }
    __syncthreads();
    // gh = h @ w_hh0.T  (WMMA, 8 waves x 3 tiles x 4 k-steps)
    v8f c0 = v8f_zero(), c1 = v8f_zero(), c2 = v8f_zero();
#pragma unroll
    for (int kt = 0; kt < 4; ++kt) {
      v16h a = load_afrag(lds_h16, kt, lane);
      c0 = wmma_f16(a, load_bfrag(whh_packed, (wave * 3 + 0) * 4 + kt, lane), c0);
      c1 = wmma_f16(a, load_bfrag(whh_packed, (wave * 3 + 1) * 4 + kt, lane), c1);
      c2 = wmma_f16(a, load_bfrag(whh_packed, (wave * 3 + 2) * 4 + kt, lane), c2);
    }
    scatter_c(lds_gi, lds_ghn, wave * 3 + 0, hv, lrow, c0);
    scatter_c(lds_gi, lds_ghn, wave * 3 + 1, hv, lrow, c1);
    scatter_c(lds_gi, lds_ghn, wave * 3 + 2, hv, lrow, c2);
    __syncthreads();
    // gates
    for (int idx = tid; idx < 16 * Hh; idx += 256) {
      int m = idx >> 7, j = idx & 127;
      float ir  = lds_gi[m * G3H + j];
      float iz  = lds_gi[m * G3H + Hh + j];
      float inn = lds_gi[m * G3H + 2 * Hh + j];
      float hn  = lds_ghn[idx] + lds_bhn[j];
      float r = 1.f / (1.f + __expf(-ir));
      float z = 1.f / (1.f + __expf(-iz));
      float nn = tanhf(inn + r * hn);
      float h = (1.f - z) * nn + z * lds_h[idx];
      lds_h[idx] = h;
      lds_h16[idx] = (_Float16)h;
      h1_out[((size_t)(r0 + m) * Tt + t) * Hh + j] = (_Float16)h;
    }
  }
}

// ---------------------------------------------------------------------------
// GRU layer 1: h1(40000,24,128) f16 -> temporal(40000,128) f16 (last step).
// Two WMMA passes per step: gi = h1_t @ w_ih1.T, gh = h2 @ w_hh1.T
// ---------------------------------------------------------------------------
__global__ __launch_bounds__(256)
void gru_layer1_kernel(const _Float16* __restrict__ h1,
                       const float* __restrict__ b_ih1,
                       const float* __restrict__ b_hh1,
                       const unsigned* __restrict__ wih_packed,
                       const unsigned* __restrict__ whh_packed,
                       _Float16* __restrict__ temporal16) {
  __shared__ float lds_gi[16 * G3H];
  __shared__ float lds_ghn[16 * Hh];
  __shared__ float lds_h[16 * Hh];
  __shared__ __align__(16) _Float16 lds_h16[16 * Hh];
  __shared__ __align__(16) _Float16 lds_a0[16 * Hh];
  __shared__ float lds_bias[G3H];
  __shared__ float lds_bhn[Hh];

  const int tid = threadIdx.x;
  const int r0  = blockIdx.x * 16;
  const int wave = tid >> 5, lane = tid & 31, hv = lane >> 4, lrow = lane & 15;

  for (int i = tid; i < 16 * Hh; i += 256) {
    lds_h[i] = 0.f;
    lds_h16[i] = (_Float16)0.f;
  }
  for (int n = tid; n < G3H; n += 256)
    lds_bias[n] = b_ih1[n] + (n < 2 * Hh ? b_hh1[n] : 0.f);
  for (int j = tid; j < Hh; j += 256) lds_bhn[j] = b_hh1[2 * Hh + j];

  for (int t = 0; t < Tt; ++t) {
    __syncthreads();
    // stage h1_t (16 rows x 64 dwords); init gi with fused bias; zero ghn
    for (int i = tid; i < 1024; i += 256) {
      int m = i >> 6, d = i & 63;
      ((unsigned*)lds_a0)[i] =
          ((const unsigned*)(h1 + ((size_t)(r0 + m) * Tt + t) * Hh))[d];
    }
    for (int p = tid; p < 16 * G3H; p += 256)
      lds_gi[p] = lds_bias[p - (p / G3H) * G3H];
    for (int i = tid; i < 16 * Hh; i += 256) lds_ghn[i] = 0.f;
    __syncthreads();
    // pass 1: gi += h1_t @ w_ih1.T (all 24 tiles into gi)
    {
      v8f c0 = v8f_zero(), c1 = v8f_zero(), c2 = v8f_zero();
#pragma unroll
      for (int kt = 0; kt < 4; ++kt) {
        v16h a = load_afrag(lds_a0, kt, lane);
        c0 = wmma_f16(a, load_bfrag(wih_packed, (wave * 3 + 0) * 4 + kt, lane), c0);
        c1 = wmma_f16(a, load_bfrag(wih_packed, (wave * 3 + 1) * 4 + kt, lane), c1);
        c2 = wmma_f16(a, load_bfrag(wih_packed, (wave * 3 + 2) * 4 + kt, lane), c2);
      }
#pragma unroll
      for (int j = 0; j < 3; ++j) {
        v8f c = (j == 0) ? c0 : (j == 1) ? c1 : c2;
        int n = (wave * 3 + j) * 16 + lrow;
#pragma unroll
        for (int vr = 0; vr < 8; ++vr) lds_gi[(vr + hv * 8) * G3H + n] += c[vr];
      }
    }
    // pass 2: gh = h2 @ w_hh1.T (r,z into gi; n-gate into ghn)
    {
      v8f c0 = v8f_zero(), c1 = v8f_zero(), c2 = v8f_zero();
#pragma unroll
      for (int kt = 0; kt < 4; ++kt) {
        v16h a = load_afrag(lds_h16, kt, lane);
        c0 = wmma_f16(a, load_bfrag(whh_packed, (wave * 3 + 0) * 4 + kt, lane), c0);
        c1 = wmma_f16(a, load_bfrag(whh_packed, (wave * 3 + 1) * 4 + kt, lane), c1);
        c2 = wmma_f16(a, load_bfrag(whh_packed, (wave * 3 + 2) * 4 + kt, lane), c2);
      }
      scatter_c(lds_gi, lds_ghn, wave * 3 + 0, hv, lrow, c0);
      scatter_c(lds_gi, lds_ghn, wave * 3 + 1, hv, lrow, c1);
      scatter_c(lds_gi, lds_ghn, wave * 3 + 2, hv, lrow, c2);
    }
    __syncthreads();
    // gates
    for (int idx = tid; idx < 16 * Hh; idx += 256) {
      int m = idx >> 7, j = idx & 127;
      float ir  = lds_gi[m * G3H + j];
      float iz  = lds_gi[m * G3H + Hh + j];
      float inn = lds_gi[m * G3H + 2 * Hh + j];
      float hn  = lds_ghn[idx] + lds_bhn[j];
      float r = 1.f / (1.f + __expf(-ir));
      float z = 1.f / (1.f + __expf(-iz));
      float nn = tanhf(inn + r * hn);
      float h = (1.f - z) * nn + z * lds_h[idx];
      lds_h[idx] = h;
      lds_h16[idx] = (_Float16)h;
      if (t == Tt - 1) temporal16[(size_t)(r0 + m) * Hh + j] = (_Float16)h;
    }
  }
}

// ---------------------------------------------------------------------------
// wh = A(40000x128 f16) @ Wpacked(128x512) -> f32 [40000][512]
// ---------------------------------------------------------------------------
__global__ __launch_bounds__(256)
void gemm_wh_kernel(const _Float16* __restrict__ A,
                    const unsigned* __restrict__ Bp,
                    float* __restrict__ out) {
  __shared__ __align__(16) _Float16 lds_a[16 * Hh];
  const int tid = threadIdx.x;
  const int r0  = blockIdx.x * 16;
  for (int i = tid; i < 1024; i += 256) {
    int m = i >> 6, d = i & 63;
    ((unsigned*)lds_a)[i] = ((const unsigned*)(A + (size_t)(r0 + m) * Hh))[d];
  }
  __syncthreads();
  const int wave = tid >> 5, lane = tid & 31, hv = lane >> 4, lrow = lane & 15;
  v8f c[4];
#pragma unroll
  for (int j = 0; j < 4; ++j) c[j] = v8f_zero();
#pragma unroll
  for (int kt = 0; kt < 4; ++kt) {
    v16h a = load_afrag(lds_a, kt, lane);
#pragma unroll
    for (int j = 0; j < 4; ++j)
      c[j] = wmma_f16(a, load_bfrag(Bp, (wave * 4 + j) * 4 + kt, lane), c[j]);
  }
#pragma unroll
  for (int j = 0; j < 4; ++j) {
    int n = (wave * 4 + j) * 16 + lrow;
#pragma unroll
    for (int vr = 0; vr < 8; ++vr)
      out[(size_t)(r0 + vr + hv * 8) * (HEADS * Hh) + n] = c[j][vr];
  }
}

// ---------------------------------------------------------------------------
// GAT scalar stages
// ---------------------------------------------------------------------------
__global__ __launch_bounds__(256)
void alpha_kernel(const float* __restrict__ wh, const float* __restrict__ a_src,
                  const float* __restrict__ a_dst, float* __restrict__ as,
                  float* __restrict__ ad) {
  int g = blockIdx.x * 256 + threadIdx.x;
  if (g >= ROWS * HEADS) return;
  int node = g >> 2, h = g & 3;
  const float* w = wh + (size_t)node * (HEADS * Hh) + h * Hh;
  float s = 0.f, d = 0.f;
  for (int j = 0; j < Hh; ++j) {
    float v = w[j];
    s += v * a_src[h * Hh + j];
    d += v * a_dst[h * Hh + j];
  }
  as[g] = s;
  ad[g] = d;
}

__global__ __launch_bounds__(256)
void fill_u32_kernel(unsigned* __restrict__ p, unsigned v, int n) {
  int g = blockIdx.x * 256 + threadIdx.x;
  if (g < n) p[g] = v;
}

static __device__ inline void edge_decode(int i, const int* __restrict__ ei,
                                          int& s, int& d) {
  if (i < Bb * Ee) {
    int b = i / Ee;
    int e = i - b * Ee;
    s = ei[e] + b * Nn;
    d = ei[Ee + e] + b * Nn;
  } else {
    s = d = i - Bb * Ee;   // self loop
  }
}

__global__ __launch_bounds__(256)
void edge_max_kernel(const int* __restrict__ ei, const float* __restrict__ as,
                     const float* __restrict__ ad, unsigned* __restrict__ emax) {
  int i = blockIdx.x * 256 + threadIdx.x;
  if (i >= EDGE_TOTAL) return;
  int s, d;
  edge_decode(i, ei, s, d);
#pragma unroll
  for (int h = 0; h < HEADS; ++h) {
    float e = as[s * HEADS + h] + ad[d * HEADS + h];
    e = e > 0.f ? e : NEG_SLOPE * e;
    atomicMax(&emax[d * HEADS + h], fkey(e));
  }
}

__global__ __launch_bounds__(256)
void edge_exp_kernel(const int* __restrict__ ei, const float* __restrict__ as,
                     const float* __restrict__ ad, const unsigned* __restrict__ emax,
                     float* __restrict__ exb, float* __restrict__ denom) {
  int i = blockIdx.x * 256 + threadIdx.x;
  if (i >= EDGE_TOTAL) return;
  int s, d;
  edge_decode(i, ei, s, d);
#pragma unroll
  for (int h = 0; h < HEADS; ++h) {
    float e = as[s * HEADS + h] + ad[d * HEADS + h];
    e = e > 0.f ? e : NEG_SLOPE * e;
    float ex = __expf(e - funkey(emax[d * HEADS + h]));
    exb[(size_t)i * HEADS + h] = ex;
    atomicAdd(&denom[d * HEADS + h], ex);
  }
}

__global__ __launch_bounds__(256)
void edge_accum_kernel(const int* __restrict__ ei, const float* __restrict__ exb,
                       const float* __restrict__ denom, const float* __restrict__ wh,
                       float* __restrict__ accum) {
  long long g = (long long)blockIdx.x * 256 + threadIdx.x;
  if (g >= (long long)EDGE_TOTAL * Hh) return;
  int i = (int)(g >> 7);
  int j = (int)(g & 127);
  int s, d;
  edge_decode(i, ei, s, d);
#pragma unroll
  for (int h = 0; h < HEADS; ++h) {
    float alpha = exb[(size_t)i * HEADS + h] / (denom[d * HEADS + h] + 1e-16f);
    atomicAdd(&accum[(size_t)d * (HEADS * Hh) + h * Hh + j],
              alpha * wh[(size_t)s * (HEADS * Hh) + h * Hh + j]);
  }
}

__global__ __launch_bounds__(256)
void gat_finish_kernel(const float* __restrict__ accum, const float* __restrict__ bias,
                       float* __restrict__ s32, _Float16* __restrict__ s16) {
  int g = blockIdx.x * 256 + threadIdx.x;
  if (g >= ROWS * Hh) return;
  int node = g >> 7, j = g & 127;
  const float* a = accum + (size_t)node * (HEADS * Hh) + j;
  float v = 0.25f * (a[0] + a[Hh] + a[2 * Hh] + a[3 * Hh]) + bias[j];
  v = v > 0.f ? v : 0.f;
  s32[g] = v;
  s16[g] = (_Float16)v;
}

__global__ __launch_bounds__(256)
void fc_kernel(const float* __restrict__ s, const float* __restrict__ fw,
               const float* __restrict__ fb, float* __restrict__ out) {
  int g = blockIdx.x * 256 + threadIdx.x;
  if (g >= ROWS * HORIZON) return;
  int node = g / HORIZON, k = g - node * HORIZON;
  const float* sr = s + (size_t)node * Hh;
  const float* w  = fw + k * Hh;
  float acc = fb[k];
  for (int j = 0; j < Hh; ++j) acc += sr[j] * w[j];
  out[g] = acc;
}

// ---------------------------------------------------------------------------
// Host launch
// ---------------------------------------------------------------------------
static inline unsigned cdiv(long long a, long long b) {
  return (unsigned)((a + b - 1) / b);
}

extern "C" void kernel_launch(void* const* d_in, const int* in_sizes, int n_in,
                              void* d_out, int out_size, void* d_ws, size_t ws_size,
                              hipStream_t stream) {
  (void)in_sizes; (void)n_in; (void)out_size; (void)ws_size;

  const float* x     = (const float*)d_in[0];
  const float* w_ih0 = (const float*)d_in[1];
  const float* w_hh0 = (const float*)d_in[2];
  const float* b_ih0 = (const float*)d_in[3];
  const float* b_hh0 = (const float*)d_in[4];
  const float* w_ih1 = (const float*)d_in[5];
  const float* w_hh1 = (const float*)d_in[6];
  const float* b_ih1 = (const float*)d_in[7];
  const float* b_hh1 = (const float*)d_in[8];
  const float* g1w   = (const float*)d_in[9];
  const float* g1as  = (const float*)d_in[10];
  const float* g1ad  = (const float*)d_in[11];
  const float* g1b   = (const float*)d_in[12];
  const float* g2w   = (const float*)d_in[13];
  const float* g2as  = (const float*)d_in[14];
  const float* g2ad  = (const float*)d_in[15];
  const float* g2b   = (const float*)d_in[16];
  const float* fcw   = (const float*)d_in[17];
  const float* fcb   = (const float*)d_in[18];
  const int*   eidx  = (const int*)d_in[19];
  float* out = (float*)d_out;

  char* ws = (char*)d_ws;
  size_t off = 0;
  auto alloc = [&](size_t bytes) -> char* {
    char* p = ws + off;
    off = (off + bytes + 255) & ~(size_t)255;
    return p;
  };

  unsigned* p_whh0 = (unsigned*)alloc((size_t)24 * 4 * 1024); // 96 frags * 1KB
  unsigned* p_wih1 = (unsigned*)alloc((size_t)24 * 4 * 1024);
  unsigned* p_whh1 = (unsigned*)alloc((size_t)24 * 4 * 1024);
  unsigned* p_g1w  = (unsigned*)alloc((size_t)32 * 4 * 1024);
  unsigned* p_g2w  = (unsigned*)alloc((size_t)32 * 4 * 1024);
  _Float16* h1     = (_Float16*)alloc((size_t)ROWS * Tt * Hh * 2);
  _Float16* temp16 = (_Float16*)alloc((size_t)ROWS * Hh * 2);
  float*    wh     = (float*)alloc((size_t)ROWS * HEADS * Hh * 4);
  float*    accum  = (float*)alloc((size_t)ROWS * HEADS * Hh * 4);
  float*    as_    = (float*)alloc((size_t)ROWS * HEADS * 4);
  float*    ad_    = (float*)alloc((size_t)ROWS * HEADS * 4);
  unsigned* emax   = (unsigned*)alloc((size_t)ROWS * HEADS * 4);
  float*    denom  = (float*)alloc((size_t)ROWS * HEADS * 4);
  float*    exb    = (float*)alloc((size_t)EDGE_TOTAL * HEADS * 4);
  float*    s32    = (float*)alloc((size_t)ROWS * Hh * 4);
  _Float16* s16    = (_Float16*)alloc((size_t)ROWS * Hh * 2);

  // 1) pack all WMMA B-operand weights (f32 -> f16 fragments)
  pack_b_kernel<<<cdiv(24 * 4 * 32, 256), 256, 0, stream>>>(w_hh0, p_whh0, 24, 4, Hh);
  pack_b_kernel<<<cdiv(24 * 4 * 32, 256), 256, 0, stream>>>(w_ih1, p_wih1, 24, 4, Hh);
  pack_b_kernel<<<cdiv(24 * 4 * 32, 256), 256, 0, stream>>>(w_hh1, p_whh1, 24, 4, Hh);
  pack_b_kernel<<<cdiv(32 * 4 * 32, 256), 256, 0, stream>>>(g1w, p_g1w, 32, 4, Hh);
  pack_b_kernel<<<cdiv(32 * 4 * 32, 256), 256, 0, stream>>>(g2w, p_g2w, 32, 4, Hh);

  // 2) GRU stack (WMMA recurrence, 16 rows per block)
  gru_layer0_kernel<<<ROWS / 16, 256, 0, stream>>>(x, w_ih0, b_ih0, b_hh0, p_whh0, h1);
  gru_layer1_kernel<<<ROWS / 16, 256, 0, stream>>>(h1, b_ih1, b_hh1, p_wih1, p_whh1,
                                                   temp16);

  // 3) two GAT passes + final FC
  const unsigned KEY_NEG_INF = 0x007FFFFFu;  // fkey(-inf)
  auto gat_pass = [&](const _Float16* in16, const unsigned* pw, const float* asw,
                      const float* adw, const float* bias) {
    fill_u32_kernel<<<cdiv(ROWS * HEADS, 256), 256, 0, stream>>>(emax, KEY_NEG_INF,
                                                                 ROWS * HEADS);
    fill_u32_kernel<<<cdiv(ROWS * HEADS, 256), 256, 0, stream>>>((unsigned*)denom, 0u,
                                                                 ROWS * HEADS);
    fill_u32_kernel<<<cdiv((long long)ROWS * HEADS * Hh, 256), 256, 0, stream>>>(
        (unsigned*)accum, 0u, ROWS * HEADS * Hh);
    gemm_wh_kernel<<<ROWS / 16, 256, 0, stream>>>(in16, pw, wh);
    alpha_kernel<<<cdiv(ROWS * HEADS, 256), 256, 0, stream>>>(wh, asw, adw, as_, ad_);
    edge_max_kernel<<<cdiv(EDGE_TOTAL, 256), 256, 0, stream>>>(eidx, as_, ad_, emax);
    edge_exp_kernel<<<cdiv(EDGE_TOTAL, 256), 256, 0, stream>>>(eidx, as_, ad_, emax,
                                                               exb, denom);
    edge_accum_kernel<<<cdiv((long long)EDGE_TOTAL * Hh, 256), 256, 0, stream>>>(
        eidx, exb, denom, wh, accum);
    gat_finish_kernel<<<cdiv(ROWS * Hh, 256), 256, 0, stream>>>(accum, bias, s32, s16);
  };
  gat_pass(temp16, p_g1w, g1as, g1ad, g1b);
  gat_pass(s16, p_g2w, g2as, g2ad, g2b);

  fc_kernel<<<cdiv(ROWS * HORIZON, 256), 256, 0, stream>>>(s32, fcw, fcb, out);
}
